// Transformer_47863115546804
// MI455X (gfx1250) — matrix operand order repaired and checked by reading, not compiled, not run on hardware
//
#include <hip/hip_runtime.h>
#include <hip/hip_bf16.h>

// ---------------------------------------------------------------------------
// Types for WMMA (CDNA5 / gfx1250, wave32)
// ---------------------------------------------------------------------------
typedef _Float16 h8  __attribute__((ext_vector_type(8)));
typedef _Float16 h16 __attribute__((ext_vector_type(16)));
typedef float    f8  __attribute__((ext_vector_type(8)));

__device__ __forceinline__ f8 wmma16(h16 a, h16 b, f8 c) {
  // D = A(16x32 f16) * B(32x16 f16) + C(16x16 f32)
  return __builtin_amdgcn_wmma_f32_16x16x32_f16(
      /*neg_a=*/false, a, /*neg_b=*/false, b,
      /*c_mod=*/(short)0, c, /*reuse_a=*/false, /*reuse_b=*/false);
}

// load 8 consecutive f32 from global, convert to f16x8
__device__ __forceinline__ h8 cvt8(const float* __restrict__ p) {
  float4 a = *(const float4*)p;
  float4 b = *(const float4*)(p + 4);
  h8 r;
  r[0] = (_Float16)a.x; r[1] = (_Float16)a.y; r[2] = (_Float16)a.z; r[3] = (_Float16)a.w;
  r[4] = (_Float16)b.x; r[5] = (_Float16)b.y; r[6] = (_Float16)b.z; r[7] = (_Float16)b.w;
  return r;
}

__device__ __forceinline__ h16 cat16(h8 x, h8 y) {
  return __builtin_shufflevector(x, y, 0,1,2,3,4,5,6,7,8,9,10,11,12,13,14,15);
}

// LDS fragment: two contiguous 8-half chunks separated by off2 halfs
__device__ __forceinline__ h16 ldfrag(const _Float16* p, int off2) {
  h8 x = *(const h8*)p;
  h8 y = *(const h8*)(p + off2);
  return cat16(x, y);
}

__device__ __forceinline__ float wred_sum(float v) {
  #pragma unroll
  for (int o = 16; o > 0; o >>= 1) v += __shfl_xor(v, o, 32);
  return v;
}
__device__ __forceinline__ float wred_max(float v) {
  #pragma unroll
  for (int o = 16; o > 0; o >>= 1) v = fmaxf(v, __shfl_xor(v, o, 32));
  return v;
}

// ---------------------------------------------------------------------------
// Problem constants
// ---------------------------------------------------------------------------
#define BB   8
#define TT   512
#define AA   512
#define DD   256
#define NH   8
#define DH   32
#define MROWS 4096   // B*T == B*A

// ---------------------------------------------------------------------------
// GEMM: C[M,N] = act( A[M,K] @ W[K,N] + bias )   (M%64==0, N%128==0, K%32==0)
// 256 threads = 8 waves; block tile 64x128; wave tile 32x32 (2x2 WMMA)
// ---------------------------------------------------------------------------
#define ASTR 40
#define WSTR 40

__global__ __launch_bounds__(256) void gemm_kernel(
    const float* __restrict__ A, const float* __restrict__ W,
    const float* __restrict__ bias, float* __restrict__ C,
    int N, int K, int act)
{
  __shared__ _Float16 Ash[64 * ASTR];
  __shared__ _Float16 Wsh[128 * WSTR];
  int tid = threadIdx.x, lane = tid & 31, wid = tid >> 5;
  int wm = wid >> 2, wn = wid & 3;
  int m0 = blockIdx.y * 64, n0 = blockIdx.x * 128;
  int mrow = lane & 15;
  int kb   = (lane < 16) ? 0 : 8;
  int koff = (lane < 16) ? 0 : 16;
  f8 acc[2][2] = {};

  for (int k0 = 0; k0 < K; k0 += 32) {
    #pragma unroll
    for (int i = 0; i < 8; ++i) {          // stage A tile 64x32
      int idx = i * 256 + tid;
      int r = idx >> 5, c = idx & 31;
      Ash[r * ASTR + c] = (_Float16)A[(size_t)(m0 + r) * K + k0 + c];
    }
    #pragma unroll
    for (int i = 0; i < 16; ++i) {         // stage W tile 32x128 (transposed)
      int idx = i * 256 + tid;
      int kk = idx >> 7, nn = idx & 127;
      Wsh[nn * WSTR + kk] = (_Float16)W[(size_t)(k0 + kk) * N + n0 + nn];
    }
    __syncthreads();

    h16 af[2], bf[2];
    #pragma unroll
    for (int mi = 0; mi < 2; ++mi)
      af[mi] = ldfrag(Ash + (wm * 32 + mi * 16 + mrow) * ASTR + kb, 16);
    #pragma unroll
    for (int ni = 0; ni < 2; ++ni)
      bf[ni] = ldfrag(Wsh + (wn * 32 + ni * 16 + mrow) * WSTR + kb, 16);
    #pragma unroll
    for (int mi = 0; mi < 2; ++mi)
      #pragma unroll
      for (int ni = 0; ni < 2; ++ni)
        acc[mi][ni] = wmma16(af[mi], bf[ni], acc[mi][ni]);
    __syncthreads();
  }

  #pragma unroll
  for (int mi = 0; mi < 2; ++mi) {
    #pragma unroll
    for (int ni = 0; ni < 2; ++ni) {
      int n = n0 + wn * 32 + ni * 16 + mrow;
      float bv = bias ? bias[n] : 0.0f;
      #pragma unroll
      for (int j = 0; j < 8; ++j) {
        int m = m0 + wm * 32 + mi * 16 + ((lane < 16) ? j : 8 + j);
        float vv = acc[mi][ni][j] + bv;
        if (act) vv = fmaxf(vv, 0.0f);
        C[(size_t)m * N + n] = vv;
      }
    }
  }
}

// ---------------------------------------------------------------------------
// Batched NT GEMM: C[z] = A[z](M x K) @ B[z](N x K)^T   (final scores)
// ---------------------------------------------------------------------------
__global__ __launch_bounds__(256) void gemm_nt_kernel(
    const float* __restrict__ A, const float* __restrict__ Bm,
    float* __restrict__ C, int Mb, int Nb, int K)
{
  __shared__ _Float16 Ash[64 * ASTR];
  __shared__ _Float16 Wsh[128 * WSTR];
  const float* Az = A  + (size_t)blockIdx.z * Mb * K;
  const float* Bz = Bm + (size_t)blockIdx.z * Nb * K;
  float*       Cz = C  + (size_t)blockIdx.z * Mb * Nb;
  int tid = threadIdx.x, lane = tid & 31, wid = tid >> 5;
  int wm = wid >> 2, wn = wid & 3;
  int m0 = blockIdx.y * 64, n0 = blockIdx.x * 128;
  int mrow = lane & 15;
  int kb   = (lane < 16) ? 0 : 8;
  int koff = (lane < 16) ? 0 : 16;
  f8 acc[2][2] = {};

  for (int k0 = 0; k0 < K; k0 += 32) {
    #pragma unroll
    for (int i = 0; i < 8; ++i) {
      int idx = i * 256 + tid;
      int r = idx >> 5, c = idx & 31;
      Ash[r * ASTR + c] = (_Float16)Az[(size_t)(m0 + r) * K + k0 + c];
    }
    #pragma unroll
    for (int i = 0; i < 16; ++i) {         // B is [N,K] row-major: direct [n][k]
      int idx = i * 256 + tid;
      int nn = idx >> 5, kk = idx & 31;
      Wsh[nn * WSTR + kk] = (_Float16)Bz[(size_t)(n0 + nn) * K + k0 + kk];
    }
    __syncthreads();

    h16 af[2], bf[2];
    #pragma unroll
    for (int mi = 0; mi < 2; ++mi)
      af[mi] = ldfrag(Ash + (wm * 32 + mi * 16 + mrow) * ASTR + kb, 16);
    #pragma unroll
    for (int ni = 0; ni < 2; ++ni)
      bf[ni] = ldfrag(Wsh + (wn * 32 + ni * 16 + mrow) * WSTR + kb, 16);
    #pragma unroll
    for (int mi = 0; mi < 2; ++mi)
      #pragma unroll
      for (int ni = 0; ni < 2; ++ni)
        acc[mi][ni] = wmma16(af[mi], bf[ni], acc[mi][ni]);
    __syncthreads();
  }

  #pragma unroll
  for (int mi = 0; mi < 2; ++mi)
    #pragma unroll
    for (int ni = 0; ni < 2; ++ni) {
      int n = n0 + wn * 32 + ni * 16 + mrow;
      #pragma unroll
      for (int j = 0; j < 8; ++j) {
        int m = m0 + wm * 32 + mi * 16 + ((lane < 16) ? j : 8 + j);
        Cz[(size_t)m * Nb + n] = acc[mi][ni][j];
      }
    }
}

// ---------------------------------------------------------------------------
// Attention (per reference's contiguous head-reshape):
//  Q,K,V,O: [B*NH][512][32] contiguous. S = softmax(mask(QK^T/sqrt(32))) @ V.
//  mQ: per-(b,q) mask or null; mK: per-(b,key) mask or null.
//  grid (T/64, B*NH), block 128 (4 waves, wave = 16 query rows)
// ---------------------------------------------------------------------------
#define SSTR 520

__global__ __launch_bounds__(128) void attn_kernel(
    const float* __restrict__ Q, const float* __restrict__ K,
    const float* __restrict__ V, float* __restrict__ O,
    const unsigned char* __restrict__ mQ, const unsigned char* __restrict__ mK)
{
  __shared__ _Float16 Vt[32 * SSTR];        // V transposed [d][key] f16
  __shared__ _Float16 Sb[4 * 16 * SSTR];    // per-wave 16x512 scores f16
  int tid = threadIdx.x, lane = tid & 31, w = tid >> 5;
  int bh = blockIdx.y, b = bh >> 3;
  const float* Qg = Q + (size_t)bh * 16384;
  const float* Kg = K + (size_t)bh * 16384;
  const float* Vg = V + (size_t)bh * 16384;
  float*       Og = O + (size_t)bh * 16384;

  for (int i = tid; i < 32 * 512; i += 128) {   // stage V^T
    int d = i >> 9, key = i & 511;
    Vt[d * SSTR + key] = (_Float16)Vg[key * 32 + d];
  }
  __syncthreads();

  int q0   = blockIdx.x * 64 + w * 16;
  int mrow = lane & 15;
  int kb   = (lane < 16) ? 0 : 8;
  int koff = (lane < 16) ? 0 : 16;
  const float scale = 0.17677669529663687f;     // 1/sqrt(32)

  const float* qp = Qg + (size_t)(q0 + mrow) * 32;
  h16 afq = cat16(cvt8(qp + kb), cvt8(qp + kb + 16));
  _Float16* Srow = Sb + w * 16 * SSTR;

  // S = Q K^T (32 key tiles of 16)
  for (int kt = 0; kt < 32; ++kt) {
    int key = kt * 16 + mrow;
    const float* kp = Kg + (size_t)key * 32 + kb;
    h16 bf = cat16(cvt8(kp), cvt8(kp + 16));
    f8 s = {};
    s = wmma16(afq, bf, s);
    bool km = mK ? (mK[b * 512 + key] != 0) : false;
    #pragma unroll
    for (int j = 0; j < 8; ++j) {
      int r = (lane < 16) ? j : 8 + j;
      bool m = km;
      if (mQ) m = m || (mQ[b * 512 + q0 + r] != 0);
      float val = m ? -30000.0f : s[j] * scale;
      Srow[r * SSTR + key] = (_Float16)val;
    }
  }
  __syncthreads();

  // row softmax (lanes 0..15 each own one row; sentinel -> 0)
  if (lane < 16) {
    const h8* pr = (const h8*)(Srow + lane * SSTR);
    float mx = -3.0e38f;
    for (int c = 0; c < 64; ++c) {
      h8 vv = pr[c];
      #pragma unroll
      for (int j = 0; j < 8; ++j) mx = fmaxf(mx, (float)vv[j]);
    }
    float sum = 0.0f;
    for (int c = 0; c < 64; ++c) {
      h8 vv = pr[c];
      #pragma unroll
      for (int j = 0; j < 8; ++j) sum += __expf((float)vv[j] - mx);
    }
    float inv = 1.0f / sum;
    h8* pw = (h8*)(Srow + lane * SSTR);
    for (int c = 0; c < 64; ++c) {
      h8 vv = pw[c], o;
      #pragma unroll
      for (int j = 0; j < 8; ++j) {
        float x = (float)vv[j];
        float p = (x <= -20000.0f) ? 0.0f : __expf(x - mx) * inv;
        o[j] = (_Float16)p;
      }
      pw[c] = o;
    }
  }
  __syncthreads();

  // O = P @ V (K = 512 in 16 steps of 32; two 16-wide output column groups)
  #pragma unroll
  for (int ni = 0; ni < 2; ++ni) {
    f8 acc = {};
    for (int kt = 0; kt < 16; ++kt) {
      const _Float16* ap = Srow + mrow * SSTR + kt * 32 + kb;
      h16 af = ldfrag(ap, 16);
      const _Float16* bp = Vt + (ni * 16 + mrow) * SSTR + kt * 32 + kb;
      h16 bf = ldfrag(bp, 16);
      acc = wmma16(af, bf, acc);
    }
    #pragma unroll
    for (int j = 0; j < 8; ++j) {
      int r = (lane < 16) ? j : 8 + j;
      Og[(size_t)(q0 + r) * 32 + ni * 16 + mrow] = acc[j];
    }
  }
}

// ---------------------------------------------------------------------------
// LayerNorm over D=256: out = LN(x (+res)) * g + be. One wave per row.
// ---------------------------------------------------------------------------
__global__ __launch_bounds__(128) void ln_kernel(
    float* __restrict__ out, const float* __restrict__ x,
    const float* __restrict__ res, const float* __restrict__ g,
    const float* __restrict__ be)
{
  int w = threadIdx.x >> 5, lane = threadIdx.x & 31;
  size_t row = (size_t)blockIdx.x * 4 + w;
  const float* xp = x + row * 256;
  const float* rp = res ? res + row * 256 : nullptr;
  float v[8], s = 0.0f, s2 = 0.0f;
  #pragma unroll
  for (int i = 0; i < 8; ++i) {
    int d = lane + i * 32;
    v[i] = xp[d] + (rp ? rp[d] : 0.0f);
    s += v[i]; s2 += v[i] * v[i];
  }
  s  = wred_sum(s);
  s2 = wred_sum(s2);
  float mean = s * (1.0f / 256.0f);
  float var  = s2 * (1.0f / 256.0f) - mean * mean;
  float rstd = rsqrtf(var + 1e-5f);
  float* op = out + row * 256;
  #pragma unroll
  for (int i = 0; i < 8; ++i) {
    int d = lane + i * 32;
    op[d] = (v[i] - mean) * rstd * g[d] + be[d];
  }
}

// ---------------------------------------------------------------------------
// Embedding sum: E[row,d] = sum_f table[f, 1 + x[row,f], d]
// ---------------------------------------------------------------------------
__global__ void embed_kernel(float* __restrict__ E, const int* __restrict__ x,
                             const float* __restrict__ table, int nf)
{
  int i = blockIdx.x * 256 + threadIdx.x;   // rows*256 threads
  int row = i >> 8, d = i & 255;
  float s = 0.0f;
  for (int f = 0; f < nf; ++f) {
    int idx = 1 + x[row * nf + f];
    s += table[((size_t)f * 33 + idx) * 256 + d];
  }
  E[i] = s;
}

// ---------------------------------------------------------------------------
// Mask kernels
// ---------------------------------------------------------------------------
__global__ void mask_pad_kernel(unsigned char* __restrict__ padK,
                                const int* __restrict__ tasks)
{
  int i = blockIdx.x * 256 + threadIdx.x;   // B*T
  const int* tp = tasks + (size_t)i * 4;
  padK[i] = (tp[0] == -1 && tp[1] == -1 && tp[2] == -1 && tp[3] == -1) ? 1 : 0;
}

__global__ void mask_scatter_kernel(unsigned char* __restrict__ assignT,
                                    const int* __restrict__ indices)
{
  int i = blockIdx.x * 256 + threadIdx.x;   // B*A*C
  int b = i / (512 * 8);
  int v = indices[i];
  if (v >= 0) assignT[b * 512 + v] = 1;
}

__global__ void mask_combine_kernel(unsigned char* __restrict__ qm,
                                    unsigned char* __restrict__ fullA,
                                    const unsigned char* __restrict__ assignT,
                                    const unsigned char* __restrict__ padK,
                                    const int* __restrict__ indices)
{
  int i = blockIdx.x * 256 + threadIdx.x;   // B*512
  qm[i] = (unsigned char)(assignT[i] | padK[i]);
  const int* ip = indices + (size_t)i * 8;
  int full = 1;
  #pragma unroll
  for (int c = 0; c < 8; ++c) full &= (ip[c] != -1) ? 1 : 0;
  fullA[i] = (unsigned char)full;
}

// ---------------------------------------------------------------------------
// Final: probs = 8*tanh(scores/16); masked softmax over a; out[b][a][t] = p
// One wave per (b,t) row.
// ---------------------------------------------------------------------------
__global__ __launch_bounds__(128) void score_softmax_kernel(
    const float* __restrict__ S, const unsigned char* __restrict__ qm,
    const unsigned char* __restrict__ fullA, float* __restrict__ out)
{
  int w = threadIdx.x >> 5, lane = threadIdx.x & 31;
  int row = blockIdx.x * 4 + w;             // row in [0, B*T)
  int b = row >> 9, t = row & 511;
  const float* sp = S + (size_t)row * 512;
  bool qmask = qm[row] != 0;
  unsigned mbits = 0;
  float vals[16], mx = -3.0e38f;
  #pragma unroll
  for (int i = 0; i < 16; ++i) {
    int a = i * 32 + lane;
    bool m = qmask || (fullA[b * 512 + a] != 0);
    float pv = 8.0f * tanhf(sp[a] * 0.0625f);
    float vv = m ? -1.0e9f : pv;
    if (m) mbits |= (1u << i);
    vals[i] = vv;
    mx = fmaxf(mx, vv);
  }
  mx = wred_max(mx);
  float sum = 0.0f;
  #pragma unroll
  for (int i = 0; i < 16; ++i) sum += __expf(vals[i] - mx);
  sum = wred_sum(sum);
  float inv = 1.0f / sum;
  #pragma unroll
  for (int i = 0; i < 16; ++i) {
    int a = i * 32 + lane;
    float p = ((mbits >> i) & 1) ? 0.0f : __expf(vals[i] - mx) * inv;
    out[((size_t)b * 512 + a) * 512 + t] = p;
  }
}

// ---------------------------------------------------------------------------
// Host-side orchestration
// ---------------------------------------------------------------------------
static inline void launch_gemm(hipStream_t s, const float* A, const float* W,
                               const float* bias, float* C, int M, int N, int K,
                               int act)
{
  dim3 g(N / 128, M / 64);
  gemm_kernel<<<g, 256, 0, s>>>(A, W, bias, C, N, K, act);
}

static void run_encoder(hipStream_t s, float* x, const float* const* p,
                        float* q, float* k, float* v, float* attn,
                        float* t1, float* t2, float* ff,
                        const unsigned char* padK)
{
  for (int i = 0; i < 3; ++i) {
    const float* wq  = p[0]  + (size_t)i * 65536;
    const float* wk  = p[1]  + (size_t)i * 65536;
    const float* wv  = p[2]  + (size_t)i * 65536;
    const float* wo  = p[3]  + (size_t)i * 65536;
    const float* w1  = p[4]  + (size_t)i * 262144;
    const float* b1  = p[5]  + (size_t)i * 1024;
    const float* w2  = p[6]  + (size_t)i * 262144;
    const float* b2  = p[7]  + (size_t)i * 256;
    const float* g1  = p[8]  + (size_t)i * 256;
    const float* be1 = p[9]  + (size_t)i * 256;
    const float* g2  = p[10] + (size_t)i * 256;
    const float* be2 = p[11] + (size_t)i * 256;

    launch_gemm(s, x, wq, nullptr, q, MROWS, 256, 256, 0);
    launch_gemm(s, x, wk, nullptr, k, MROWS, 256, 256, 0);
    launch_gemm(s, x, wv, nullptr, v, MROWS, 256, 256, 0);
    attn_kernel<<<dim3(8, 64), 128, 0, s>>>(q, k, v, attn, nullptr, padK);
    launch_gemm(s, attn, wo, nullptr, t1, MROWS, 256, 256, 0);
    ln_kernel<<<1024, 128, 0, s>>>(t2, x, t1, g1, be1);
    launch_gemm(s, t2, w1, b1, ff, MROWS, 1024, 256, 1);
    launch_gemm(s, ff, w2, b2, t1, MROWS, 256, 1024, 0);
    ln_kernel<<<1024, 128, 0, s>>>(x, t2, t1, g2, be2);
  }
}

extern "C" void kernel_launch(void* const* d_in, const int* in_sizes, int n_in,
                              void* d_out, int out_size, void* d_ws, size_t ws_size,
                              hipStream_t stream)
{
  (void)in_sizes; (void)n_in; (void)out_size; (void)ws_size;

  const int*   tasks    = (const int*)  d_in[0];
  const int*   agents   = (const int*)  d_in[1];
  const int*   indices  = (const int*)  d_in[2];
  const float* emb_t    = (const float*)d_in[3];
  const float* emb_t_wo = (const float*)d_in[4];
  const float* emb_t_bo = (const float*)d_in[5];
  const float* emb_a    = (const float*)d_in[6];
  const float* emb_a_wo = (const float*)d_in[7];
  const float* emb_a_bo = (const float*)d_in[8];
  const float* et[12]; for (int i = 0; i < 12; ++i) et[i] = (const float*)d_in[9 + i];
  const float* ea[12]; for (int i = 0; i < 12; ++i) ea[i] = (const float*)d_in[21 + i];
  const float* de_wq  = (const float*)d_in[33];
  const float* de_wk  = (const float*)d_in[34];
  const float* de_wv  = (const float*)d_in[35];
  const float* de_wo  = (const float*)d_in[36];
  const float* de_w1  = (const float*)d_in[37];
  const float* de_b1  = (const float*)d_in[38];
  const float* de_w2  = (const float*)d_in[39];
  const float* de_b2  = (const float*)d_in[40];
  const float* de_g1  = (const float*)d_in[41];
  const float* de_be1 = (const float*)d_in[42];
  const float* de_g2  = (const float*)d_in[43];
  const float* de_be2 = (const float*)d_in[44];
  const float* do_wq  = (const float*)d_in[45];
  const float* do_wk  = (const float*)d_in[46];

  const size_t NE = (size_t)MROWS * DD;     // 1,048,576 floats
  float* F    = (float*)d_ws;
  float* xt   = F;
  float* xa   = F + 1 * NE;
  float* q    = F + 2 * NE;
  float* k    = F + 3 * NE;
  float* v    = F + 4 * NE;
  float* attn = F + 5 * NE;
  float* t1   = F + 6 * NE;
  float* t2   = F + 7 * NE;
  float* ff   = F + 8 * NE;                 // 4*NE floats (FF=1024); also scores
  unsigned char* U       = (unsigned char*)(F + 12 * NE);
  unsigned char* padK    = U;
  unsigned char* assignT = U + 4096;
  unsigned char* qmArr   = U + 8192;
  unsigned char* fullA   = U + 12288;

  // ---- masks ----
  mask_pad_kernel<<<16, 256, 0, stream>>>(padK, tasks);
  hipMemsetAsync(assignT, 0, 4096, stream);
  mask_scatter_kernel<<<128, 256, 0, stream>>>(assignT, indices);
  mask_combine_kernel<<<16, 256, 0, stream>>>(qmArr, fullA, assignT, padK, indices);

  // ---- embeddings ----
  embed_kernel<<<4096, 256, 0, stream>>>(t1, tasks, emb_t, 4);
  launch_gemm(stream, t1, emb_t_wo, emb_t_bo, xt, MROWS, 256, 256, 0);
  embed_kernel<<<4096, 256, 0, stream>>>(t1, agents, emb_a, 2);
  launch_gemm(stream, t1, emb_a_wo, emb_a_bo, xa, MROWS, 256, 256, 0);

  // ---- encoders ----
  run_encoder(stream, xt, et, q, k, v, attn, t1, t2, ff, padK);
  run_encoder(stream, xa, ea, q, k, v, attn, t1, t2, ff, padK);

  // ---- decoder layer ----
  launch_gemm(stream, xt, de_wq, nullptr, q, MROWS, 256, 256, 0);
  launch_gemm(stream, xa, de_wk, nullptr, k, MROWS, 256, 256, 0);
  launch_gemm(stream, xa, de_wv, nullptr, v, MROWS, 256, 256, 0);
  attn_kernel<<<dim3(8, 64), 128, 0, stream>>>(q, k, v, attn, qmArr, fullA);
  launch_gemm(stream, attn, de_wo, nullptr, t1, MROWS, 256, 256, 0);
  ln_kernel<<<1024, 128, 0, stream>>>(t2, xt, t1, de_g1, de_be1);
  launch_gemm(stream, t2, de_w1, de_b1, ff, MROWS, 512, 256, 1);
  launch_gemm(stream, ff, de_w2, de_b2, t1, MROWS, 256, 512, 0);
  ln_kernel<<<1024, 128, 0, stream>>>(attn, t2, t1, de_g2, de_be2);  // attn := h

  // ---- final scores ----
  launch_gemm(stream, attn, do_wq, nullptr, q, MROWS, 256, 256, 0);  // hq
  launch_gemm(stream, xa,   do_wk, nullptr, k, MROWS, 256, 256, 0);  // hk
  gemm_nt_kernel<<<dim3(4, 8, BB), 256, 0, stream>>>(q, k, ff, 512, 512, 256);
  score_softmax_kernel<<<1024, 128, 0, stream>>>(ff, qmArr, fullA, (float*)d_out);
}
